// AdaptiveAttention_36309653520521
// MI455X (gfx1250) — matrix operand order repaired and checked by reading, not compiled, not run on hardware
//
#include <hip/hip_runtime.h>

// Problem constants (fixed by the reference harness).
#define N_  2
#define L_  2048
#define S_  2048
#define H_  16
#define E_  64
#define D_  64
#define NEG_  (-1e9f)
#define TEMP_ (0.125f)            /* 1/sqrt(E) */
#define NEGT_ (NEG_ * TEMP_)      /* mask folded through softmax temp */

typedef __attribute__((ext_vector_type(16))) __bf16          v16bf;
typedef __attribute__((ext_vector_type(16))) unsigned short  v16us;
typedef __attribute__((ext_vector_type(8)))  unsigned short  v8us;
typedef __attribute__((ext_vector_type(8)))  float           v8f;
typedef __attribute__((ext_vector_type(4)))  float           v4f;

// fp32 -> bf16 round-to-nearest-even
static __device__ inline unsigned short f2bf(float f) {
  unsigned u = __builtin_bit_cast(unsigned, f);
  u += 0x7fffu + ((u >> 16) & 1u);
  return (unsigned short)(u >> 16);
}

// Reduce across the 16-lane half-groups (lanes 0-15 and 16-31 independently).
static __device__ inline float redmax16(float v) {
  v = fmaxf(v, __shfl_xor(v, 1, 32));
  v = fmaxf(v, __shfl_xor(v, 2, 32));
  v = fmaxf(v, __shfl_xor(v, 4, 32));
  v = fmaxf(v, __shfl_xor(v, 8, 32));
  return v;
}
static __device__ inline float redsum16(float v) {
  v += __shfl_xor(v, 1, 32);
  v += __shfl_xor(v, 2, 32);
  v += __shfl_xor(v, 4, 32);
  v += __shfl_xor(v, 8, 32);
  return v;
}

// Load a 16 x bf16 fragment (elems p[0..7] and p[16..23]) as two aligned 16B LDS reads.
static __device__ inline v16bf ld_frag16(const unsigned short* p) {
  v8us lo = *(const v8us*)p;
  v8us hi = *(const v8us*)(p + 16);
  v16us c = __builtin_shufflevector(lo, hi, 0, 1, 2, 3, 4, 5, 6, 7,
                                    8, 9, 10, 11, 12, 13, 14, 15);
  return __builtin_bit_cast(v16bf, c);
}

static __device__ inline v8f wmma_bf16(v16bf a, v16bf b, v8f c) {
  return __builtin_amdgcn_wmma_f32_16x16x32_bf16(
      /*neg_a=*/false, a, /*neg_b=*/false, b,
      /*c_mod=*/(short)0, c, /*reuse_a=*/false, /*reuse_b=*/false);
}

__global__ __launch_bounds__(256) void
fa_fwd_kernel(const float* __restrict__ q, const float* __restrict__ k,
              const float* __restrict__ v, const float* __restrict__ klm,
              const float* __restrict__ cu, float* __restrict__ out) {
  // K tile: 32 keys x 64 E, bf16, row stride 72 elems (144B, 16B aligned).
  __shared__ __align__(16) unsigned short Klds[32][72];
  // V tile transposed: 64 D x 32 keys, row stride 40 elems (80B, 16B aligned).
  __shared__ __align__(16) unsigned short Vtlds[64][40];
  // Per-wave P scratch: 16 rows x 32 cols (padded to 40).
  __shared__ __align__(16) unsigned short Plds[8][16][40];

  const int n  = blockIdx.z;
  const int h  = blockIdx.y;
  const int m0 = blockIdx.x * 128;      // 128 query rows per block

  const int tid  = threadIdx.x;
  const int wv   = tid >> 5;            // wave id 0..7 (16 query rows each)
  const int lane = tid & 31;
  const int col  = lane & 15;           // C-layout column / A-layout row
  const int hh   = lane >> 4;           // lane half

  // ---- Load Q A-fragments once (lane = row, ISA 16-bit A layout) ----
  const int mwave = m0 + 16 * wv;       // first query row of this wave
  const float* qrow =
      q + (((size_t)n * L_ + (mwave + col)) * H_ + h) * (size_t)E_;
  v16bf qfrag[2];
#pragma unroll
  for (int e0 = 0; e0 < 2; ++e0) {
    const float* g0 = qrow + 32 * e0 + 8 * hh;        // K rel 8h+0..7
    const float* g1 = g0 + 16;                        // K rel 8h+16..23
    v4f x0 = ((const v4f*)g0)[0], x1 = ((const v4f*)g0)[1];
    v4f x2 = ((const v4f*)g1)[0], x3 = ((const v4f*)g1)[1];
    v16us a;
#pragma unroll
    for (int i = 0; i < 4; ++i) {
      a[i]      = f2bf(x0[i]);
      a[4 + i]  = f2bf(x1[i]);
      a[8 + i]  = f2bf(x2[i]);
      a[12 + i] = f2bf(x3[i]);
    }
    qfrag[e0] = __builtin_bit_cast(v16bf, a);
  }

  // ---- Online softmax state ----
  float mrun[8], lsum[8];
  v8f acc[4];
#pragma unroll
  for (int r = 0; r < 8; ++r) { mrun[r] = -1e30f; lsum[r] = 0.0f; }
#pragma unroll
  for (int dt = 0; dt < 4; ++dt) acc[dt] = (v8f){0, 0, 0, 0, 0, 0, 0, 0};

  const float klmsc = TEMP_;
  const float* cub = cu + ((size_t)n * L_ + mwave) * (size_t)S_;
  unsigned short (*Pl)[40] = Plds[wv];

  const int sEnd = m0 + 128;            // causal: max key index +1 for block
  const int waveLastRow = mwave + 15;

  // Cooperative-load addressing: 256 threads, 8 floats each per 32x64 tile.
  const int sloc = tid >> 3;            // 0..31
  const int e8   = (tid & 7) << 3;      // 0,8,...,56

  for (int s0 = 0; s0 < sEnd; s0 += 32) {
    // ---- Stage K (row-major) and V (transposed) tiles into LDS ----
    {
      size_t rowoff =
          (((size_t)n * S_ + (s0 + sloc)) * H_ + h) * (size_t)E_ + e8;
      v4f k0 = *(const v4f*)(k + rowoff);
      v4f k1 = *(const v4f*)(k + rowoff + 4);
      v4f v0 = *(const v4f*)(v + rowoff);
      v4f v1 = *(const v4f*)(v + rowoff + 4);
      v8us kp;
#pragma unroll
      for (int i = 0; i < 4; ++i) { kp[i] = f2bf(k0[i]); kp[4 + i] = f2bf(k1[i]); }
      *(v8us*)&Klds[sloc][e8] = kp;
#pragma unroll
      for (int i = 0; i < 4; ++i) {
        Vtlds[e8 + i][sloc]     = f2bf(v0[i]);
        Vtlds[e8 + 4 + i][sloc] = f2bf(v1[i]);
      }
      if (s0 + 32 < sEnd) {             // gfx1250 global_prefetch_b8 of next tile
        __builtin_prefetch(k + rowoff + (size_t)32 * H_ * E_, 0, 0);
        __builtin_prefetch(v + rowoff + (size_t)32 * H_ * E_, 0, 0);
      }
    }
    __syncthreads();

    if (s0 <= waveLastRow) {            // per-wave causal skip
      // ---- QK^T: 16x32 logits tile via 4 WMMAs ----
      v8f c0 = (v8f){0, 0, 0, 0, 0, 0, 0, 0};
      v8f c1 = (v8f){0, 0, 0, 0, 0, 0, 0, 0};
#pragma unroll
      for (int e0 = 0; e0 < 2; ++e0) {
        v16bf b0 = ld_frag16(&Klds[col][32 * e0 + 8 * hh]);
        v16bf b1 = ld_frag16(&Klds[16 + col][32 * e0 + 8 * hh]);
        c0 = wmma_bf16(qfrag[e0], b0, c0);
        c1 = wmma_bf16(qfrag[e0], b1, c1);
      }

      // ---- masks + online softmax (rows: VGPR r, half hh -> row r+8*hh) ----
      const int g0c = s0 + col, g1c = g0c + 16;
      const float kl0 = klm[(size_t)n * S_ + g0c] * klmsc;
      const float kl1 = klm[(size_t)n * S_ + g1c] * klmsc;
      float corr[8];
#pragma unroll
      for (int r = 0; r < 8; ++r) {
        const int lr = r + 8 * hh;
        const int gm = mwave + lr;
        const float cu0 = cub[(size_t)lr * S_ + g0c];
        const float cu1 = cub[(size_t)lr * S_ + g1c];
        float x0 = c0[r] * (cu0 * TEMP_) + ((g0c <= gm) ? 0.f : NEGT_) + kl0;
        float x1 = c1[r] * (cu1 * TEMP_) + ((g1c <= gm) ? 0.f : NEGT_) + kl1;
        float mx   = redmax16(fmaxf(x0, x1));
        float mnew = fmaxf(mrun[r], mx);
        float cr   = __expf(mrun[r] - mnew);
        float e0v  = __expf(x0 - mnew);
        float e1v  = __expf(x1 - mnew);
        mrun[r] = mnew;
        lsum[r] = lsum[r] * cr + redsum16(e0v + e1v);
        corr[r] = cr;
        Pl[lr][col]      = f2bf(e0v);   // C-layout -> LDS bounce
        Pl[lr][col + 16] = f2bf(e1v);
      }
#pragma unroll
      for (int dt = 0; dt < 4; ++dt)
#pragma unroll
        for (int r = 0; r < 8; ++r) acc[dt][r] *= corr[r];

      // ---- P (A-layout from LDS) x V: 4 WMMAs into the output tile ----
      v8us plo = *(const v8us*)&Pl[col][8 * hh];
      v8us phi = *(const v8us*)&Pl[col][8 * hh + 16];
      v16us pc = __builtin_shufflevector(plo, phi, 0, 1, 2, 3, 4, 5, 6, 7,
                                         8, 9, 10, 11, 12, 13, 14, 15);
      v16bf ap = __builtin_bit_cast(v16bf, pc);
#pragma unroll
      for (int dt = 0; dt < 4; ++dt) {
        v16bf bv = ld_frag16(&Vtlds[dt * 16 + col][8 * hh]);
        acc[dt] = wmma_bf16(ap, bv, acc[dt]);
      }
    }
    __syncthreads();
  }

  // ---- Epilogue: normalize and store [N,L,H,D] fp32 ----
  float* ob = out + (((size_t)n * L_ + mwave) * H_ + h) * (size_t)D_;
#pragma unroll
  for (int r = 0; r < 8; ++r) {
    const int lr = r + 8 * hh;
    const float inv = 1.0f / lsum[r];
#pragma unroll
    for (int dt = 0; dt < 4; ++dt)
      ob[(size_t)lr * (H_ * D_) + dt * 16 + col] = acc[dt][r] * inv;
  }
}

extern "C" void kernel_launch(void* const* d_in, const int* in_sizes, int n_in,
                              void* d_out, int out_size, void* d_ws, size_t ws_size,
                              hipStream_t stream) {
  (void)in_sizes; (void)n_in; (void)out_size; (void)d_ws; (void)ws_size;
  const float* q   = (const float*)d_in[0];   // [N,L,H,E]
  const float* k   = (const float*)d_in[1];   // [N,S,H,E]
  const float* v   = (const float*)d_in[2];   // [N,S,H,D]
  // d_in[3] (attn_mask) is the causal tril mask; computed analytically.
  const float* klm = (const float*)d_in[4];   // [N,S]
  const float* cu  = (const float*)d_in[5];   // [N,L,S]
  float* out = (float*)d_out;                 // [N,L,H,D]

  dim3 grid(L_ / 128, H_, N_);
  dim3 block(256);
  hipLaunchKernelGGL(fa_fwd_kernel, grid, block, 0, stream, q, k, v, klm, cu, out);
}
